// LSSFPN_69638599737513
// MI455X (gfx1250) — matrix operand order repaired and checked by reading, not compiled, not run on hardware
//
#include <hip/hip_runtime.h>
#include <hip/hip_bf16.h>
#include <stdint.h>

#define DEV __device__ __forceinline__

typedef __attribute__((ext_vector_type(16))) __bf16 v16bf;
typedef __attribute__((ext_vector_type(8)))  float  v8f;

// ---------------- problem constants ----------------
static constexpr int NCAM = 6, CIN = 512, DBINS = 112, CCTX = 80;
static constexpr int FH = 16, FW = 44, NPIX = FH * FW;          // 704
static constexpr int PH = FH + 2, PW = FW + 2, PPIX = PH * PW;  // 18*46 = 828
static constexpr int KD = 512;                                  // channel dim
static constexpr int NPTS = NCAM * DBINS * NPIX;                // 473088

// ---------------- workspace layout (bytes) ----------------
static constexpr size_t SZ_XT  = (size_t)NCAM * PPIX * KD * 2;      // 5,087,232
static constexpr size_t SZ_W3  = (size_t)9 * KD * KD * 2;           // 4,718,592
static constexpr size_t OFF_XTA = 0;
static constexpr size_t OFF_XTB = OFF_XTA + SZ_XT;
static constexpr size_t OFF_WS1 = OFF_XTB + SZ_XT;
static constexpr size_t OFF_WS2 = OFF_WS1 + SZ_W3;
static constexpr size_t OFF_WD1 = OFF_WS2 + SZ_W3;
static constexpr size_t OFF_WC1 = OFF_WD1 + SZ_W3;
static constexpr size_t OFF_WD2 = OFF_WC1 + SZ_W3;                  // 112*512 bf16
static constexpr size_t OFF_WC2 = OFF_WD2 + (size_t)DBINS * KD * 2; // 80*512 bf16
static constexpr size_t OFF_LOG = OFF_WC2 + (size_t)CCTX * KD * 2;  // 6*112*704 f32
static constexpr size_t OFF_CTX = OFF_LOG + (size_t)NCAM * DBINS * NPIX * 4;
static constexpr size_t OFF_CAM = OFF_CTX + (size_t)NCAM * CCTX * NPIX * 4;
static constexpr size_t OFF_VOX = OFF_CAM + 1024;                   // NPTS ints

// ---------------- small helpers ----------------
DEV uint16_t f2bfbits(float x) {
  uint32_t u = __float_as_uint(x);
  uint32_t r = u + 0x7FFFu + ((u >> 16) & 1u);
  return (uint16_t)(r >> 16);
}
DEV __bf16 f2bf(float x) {
  uint16_t h = f2bfbits(x);
  __bf16 b; __builtin_memcpy(&b, &h, 2); return b;
}
DEV uint32_t pk2_relu(float lo, float hi) {
  return (uint32_t)f2bfbits(fmaxf(lo, 0.f)) |
         ((uint32_t)f2bfbits(fmaxf(hi, 0.f)) << 16);
}

union FragBF { v16bf v; uint4 q[2]; };

// A fragment (16x32, 16-bit): lane l holds row m0+(l&15); halves 0..7 -> K=kb+0..7,
// halves 8..15 -> K=16+kb+0..7, kb = (l>>4)*8. Storage: row-major [M][ld].
DEV v16bf load_a_frag(const __bf16* __restrict__ base, int ld, int m0, int k0) {
  int l = threadIdx.x & 31;
  const __bf16* p = base + (size_t)(m0 + (l & 15)) * ld + k0 + ((l >> 4) << 3);
  FragBF f;
  f.q[0] = *reinterpret_cast<const uint4*>(p);
  f.q[1] = *reinterpret_cast<const uint4*>(p + 16);
  return f.v;
}

// B fragment (32x16, 16-bit): lane l holds col n=l&15; halves 0..7 -> K=kb+0..7,
// halves 8..15 -> K=16+kb+0..7, kb = (l>>4)*8.
// Storage is [N][K] row-major (K contiguous); caller passes the per-lane row index.
DEV v16bf load_b_frag(const __bf16* __restrict__ base, int ld, int row, int k0) {
  int l = threadIdx.x & 31;
  const __bf16* p = base + (size_t)row * ld + k0 + ((l >> 4) << 3);
  FragBF f;
  f.q[0] = *reinterpret_cast<const uint4*>(p);
  f.q[1] = *reinterpret_cast<const uint4*>(p + 16);
  return f.v;
}

// ---------------- utility kernels ----------------
__global__ void k_zero_u4(uint4* p, long n) {
  long i = (long)blockIdx.x * blockDim.x + threadIdx.x;
  if (i < n) p[i] = make_uint4(0u, 0u, 0u, 0u);
}

// [OC][IC][3][3] f32 -> [t][OC][IC] bf16
__global__ void k_repack_w3(const float* __restrict__ w, __bf16* __restrict__ o) {
  int idx = blockIdx.x * blockDim.x + threadIdx.x;
  if (idx >= KD * KD * 9) return;
  int t = idx % 9, ic = (idx / 9) % KD, oc = idx / (9 * KD);
  o[(size_t)t * KD * KD + (size_t)oc * KD + ic] = f2bf(w[idx]);
}

__global__ void k_cvt_bf(const float* __restrict__ s, __bf16* __restrict__ d, int n) {
  int i = blockIdx.x * blockDim.x + threadIdx.x;
  if (i < n) d[i] = f2bf(s[i]);
}

// src [6][512][16][44] f32 -> Xt [6][PPIX][512] bf16 (interior, borders pre-zeroed)
__global__ void k_cvt_input(const float* __restrict__ src, __bf16* __restrict__ xt) {
  int idx = blockIdx.x * blockDim.x + threadIdx.x;
  if (idx >= NCAM * CIN * NPIX) return;
  int x = idx % FW, y = (idx / FW) % FH, c = (idx / NPIX) % CIN, img = idx / (NPIX * CIN);
  xt[(size_t)img * PPIX * KD + (size_t)((y + 1) * PW + (x + 1)) * KD + c] = f2bf(src[idx]);
}

// ---------------- WMMA conv kernels ----------------
// 3x3 conv, 512->512, relu, writes bf16 into next padded Xt buffer.
__global__ __launch_bounds__(32) void k_conv3x3_wmma(
    const __bf16* __restrict__ Xin, const __bf16* __restrict__ Wt,
    __bf16* __restrict__ Xout) {
  const int l = threadIdx.x & 31;
  const int img = blockIdx.z;
  const int m0 = blockIdx.y * 64;   // out-channel block
  const int n0 = blockIdx.x * 32;   // pixel block
  const __bf16* Xi = Xin + (size_t)img * PPIX * KD;
  __bf16* Xo = Xout + (size_t)img * PPIX * KD;

  int pidx[2];
#pragma unroll
  for (int j = 0; j < 2; j++) {
    int n = n0 + 16 * j + (l & 15);
    int y = n / FW, x = n % FW;
    pidx[j] = y * PW + x;  // top-left tap position in padded space
  }

  v8f acc[4][2];
#pragma unroll
  for (int i = 0; i < 4; i++)
#pragma unroll
    for (int j = 0; j < 2; j++)
#pragma unroll
      for (int r = 0; r < 8; r++) acc[i][j][r] = 0.f;

  for (int t = 0; t < 9; t++) {
    const __bf16* Wtap = Wt + (size_t)t * KD * KD;
    const int toff = (t / 3) * PW + (t % 3);
    for (int kb = 0; kb < KD / 32; kb++) {
      int k0 = kb * 32;
      v16bf a[4], b[2];
#pragma unroll
      for (int i = 0; i < 4; i++) a[i] = load_a_frag(Wtap, KD, m0 + 16 * i, k0);
#pragma unroll
      for (int j = 0; j < 2; j++) b[j] = load_b_frag(Xi, KD, pidx[j] + toff, k0);
#pragma unroll
      for (int i = 0; i < 4; i++)
#pragma unroll
        for (int j = 0; j < 2; j++)
          acc[i][j] = __builtin_amdgcn_wmma_f32_16x16x32_bf16(
              false, a[i], false, b[j], (short)0, acc[i][j], false, false);
    }
  }

  // C layout: lane l, vgpr r -> m = 16*i + r + 8*(l>>4); 8 consecutive out-chans
  // per lane pack into one 16B store straight into the [pix][oc] layout.
#pragma unroll
  for (int i = 0; i < 4; i++) {
    int oc0 = m0 + 16 * i + ((l >> 4) << 3);
#pragma unroll
    for (int j = 0; j < 2; j++) {
      uint4 q;
      q.x = pk2_relu(acc[i][j][0], acc[i][j][1]);
      q.y = pk2_relu(acc[i][j][2], acc[i][j][3]);
      q.z = pk2_relu(acc[i][j][4], acc[i][j][5]);
      q.w = pk2_relu(acc[i][j][6], acc[i][j][7]);
      *reinterpret_cast<uint4*>(Xo + (size_t)(pidx[j] + PW + 1) * KD + oc0) = q;
    }
  }
}

// 1x1 conv head, 512->M, + bias, f32 out [img][M][704]
__global__ __launch_bounds__(32) void k_conv1x1_wmma(
    const __bf16* __restrict__ Xin, const __bf16* __restrict__ W,
    const float* __restrict__ bias, float* __restrict__ out, int M) {
  const int l = threadIdx.x & 31;
  const int img = blockIdx.z;
  const int m0 = blockIdx.y * 16;
  const int n0 = blockIdx.x * 64;
  const __bf16* Xi = Xin + (size_t)img * PPIX * KD;
  float* O = out + (size_t)img * M * NPIX;

  int pidx[4], ncol[4];
#pragma unroll
  for (int j = 0; j < 4; j++) {
    int n = n0 + 16 * j + (l & 15);
    int y = n / FW, x = n % FW;
    pidx[j] = (y + 1) * PW + (x + 1);
    ncol[j] = n;
  }

  v8f acc[4];
#pragma unroll
  for (int j = 0; j < 4; j++)
#pragma unroll
    for (int r = 0; r < 8; r++) acc[j][r] = 0.f;

  for (int kb = 0; kb < KD / 32; kb++) {
    int k0 = kb * 32;
    v16bf a = load_a_frag(W, KD, m0, k0);
#pragma unroll
    for (int j = 0; j < 4; j++) {
      v16bf b = load_b_frag(Xi, KD, pidx[j], k0);
      acc[j] = __builtin_amdgcn_wmma_f32_16x16x32_bf16(
          false, a, false, b, (short)0, acc[j], false, false);
    }
  }

#pragma unroll
  for (int j = 0; j < 4; j++)
#pragma unroll
    for (int r = 0; r < 8; r++) {
      int oc = m0 + r + ((l >> 4) << 3);
      O[(size_t)oc * NPIX + ncol[j]] = acc[j][r] + bias[oc];
    }
}

// ---------------- softmax over depth bins ----------------
__global__ void k_softmax_d(float* __restrict__ logits) {
  int idx = blockIdx.x * blockDim.x + threadIdx.x;
  if (idx >= NCAM * NPIX) return;
  int img = idx / NPIX, pix = idx % NPIX;
  float* p = logits + (size_t)img * DBINS * NPIX + pix;
  float m = -1e30f;
  for (int d = 0; d < DBINS; d++) m = fmaxf(m, p[(size_t)d * NPIX]);
  float s = 0.f;
  for (int d = 0; d < DBINS; d++) s += __expf(p[(size_t)d * NPIX] - m);
  float inv = 1.f / s;
  for (int d = 0; d < DBINS; d++)
    p[(size_t)d * NPIX] = __expf(p[(size_t)d * NPIX] - m) * inv;
}

// ---------------- geometry ----------------
DEV void inv3(const float* a, float* o) {
  float d = a[0] * (a[4] * a[8] - a[5] * a[7]) - a[1] * (a[3] * a[8] - a[5] * a[6]) +
            a[2] * (a[3] * a[7] - a[4] * a[6]);
  float id = 1.f / d;
  o[0] = (a[4] * a[8] - a[5] * a[7]) * id; o[1] = (a[2] * a[7] - a[1] * a[8]) * id;
  o[2] = (a[1] * a[5] - a[2] * a[4]) * id; o[3] = (a[5] * a[6] - a[3] * a[8]) * id;
  o[4] = (a[0] * a[8] - a[2] * a[6]) * id; o[5] = (a[2] * a[3] - a[0] * a[5]) * id;
  o[6] = (a[3] * a[7] - a[4] * a[6]) * id; o[7] = (a[1] * a[6] - a[0] * a[7]) * id;
  o[8] = (a[0] * a[4] - a[1] * a[3]) * id;
}
DEV void inv4(const float* ain, float* o) {
  float a[4][8];
  for (int r = 0; r < 4; r++)
    for (int c = 0; c < 4; c++) { a[r][c] = ain[r * 4 + c]; a[r][4 + c] = (r == c) ? 1.f : 0.f; }
  for (int col = 0; col < 4; col++) {
    int piv = col;
    for (int r = col + 1; r < 4; r++) if (fabsf(a[r][col]) > fabsf(a[piv][col])) piv = r;
    if (piv != col)
      for (int c = 0; c < 8; c++) { float t = a[col][c]; a[col][c] = a[piv][c]; a[piv][c] = t; }
    float iv = 1.f / a[col][col];
    for (int c = 0; c < 8; c++) a[col][c] *= iv;
    for (int r = 0; r < 4; r++)
      if (r != col) { float f = a[r][col]; for (int c = 0; c < 8; c++) a[r][c] -= f * a[col][c]; }
  }
  for (int r = 0; r < 4; r++) for (int c = 0; c < 4; c++) o[r * 4 + c] = a[r][4 + c];
}

// per-camera fused matrices: cam[n*24] = {P(9), tvec(3), M(3x4)}
__global__ void k_cam_setup(const float* __restrict__ s2e, const float* __restrict__ intrin,
                            const float* __restrict__ ida, const float* __restrict__ bda,
                            float* __restrict__ cam) {
  int n = threadIdx.x;
  if (n >= NCAM) return;
  float K[9];
  for (int r = 0; r < 3; r++) for (int c = 0; c < 3; c++) K[r * 3 + c] = intrin[n * 16 + r * 4 + c];
  float Kinv[9]; inv3(K, Kinv);
  float idainv[16]; inv4(ida + n * 16, idainv);
  float* o = cam + n * 24;
  for (int i = 0; i < 3; i++) {
    for (int j = 0; j < 3; j++) {
      float s = 0.f;
      for (int k = 0; k < 3; k++) s += Kinv[i * 3 + k] * idainv[k * 4 + j];
      o[i * 3 + j] = s;
    }
    float t = 0.f;
    for (int k = 0; k < 3; k++) t += Kinv[i * 3 + k] * idainv[k * 4 + 3];
    o[9 + i] = t;
  }
  for (int i = 0; i < 3; i++)
    for (int j = 0; j < 4; j++) {
      float s = 0.f;
      for (int k = 0; k < 4; k++) s += bda[i * 4 + k] * s2e[n * 16 + k * 4 + j];
      o[12 + i * 4 + j] = s;
    }
}

__global__ void k_geom_vox(const float* __restrict__ cam, int* __restrict__ vox) {
  int idx = blockIdx.x * blockDim.x + threadIdx.x;
  if (idx >= NPTS) return;
  int pix = idx % NPIX, d = (idx / NPIX) % DBINS, n = idx / (NPIX * DBINS);
  int x = pix % FW, y = pix / FW;
  float u = x * (703.0f / 43.0f);
  float v = y * 17.0f;
  float depth = 2.0f + d * (56.0f / 111.0f);
  const float* c = cam + n * 24;
  float rx = c[0] * u + c[1] * v + c[2] + c[9];
  float ry = c[3] * u + c[4] * v + c[5] + c[10];
  float rz = c[6] * u + c[7] * v + c[8] + c[11];
  float px = rx * depth, py = ry * depth, pz = rz * depth;
  float ex = c[12] * px + c[13] * py + c[14] * pz + c[15];
  float ey = c[16] * px + c[17] * py + c[18] * pz + c[19];
  float ez = c[20] * px + c[21] * py + c[22] * pz + c[23];
  // match reference: int32 truncation toward zero, then bounds check on ints
  float fgx = truncf((ex + 51.2f) * 1.25f);
  float fgy = truncf((ey + 51.2f) * 1.25f);
  float fgz = truncf((ez + 5.0f) * 0.125f);
  bool valid = (fgx >= 0.f) & (fgx < 128.f) & (fgy >= 0.f) & (fgy < 128.f) &
               (fgz >= 0.f) & (fgz < 1.f);
  vox[idx] = valid ? ((int)fgy * 128 + (int)fgx) : -1;
}

// scatter: out[c, vox] += prob[n,d,pix] * ctx[n,c,pix]; 3 points x 80 chans / block
__global__ void k_scatter(const float* __restrict__ prob, const float* __restrict__ ctx,
                          const int* __restrict__ vox, float* __restrict__ out) {
  int t = threadIdx.x;
  if (t >= 240) return;
  int pt = blockIdx.x * 3 + t / CCTX;
  int c = t % CCTX;
  if (pt >= NPTS) return;
  int v = vox[pt];
  if (v < 0) return;
  int pix = pt % NPIX, d = (pt / NPIX) % DBINS, n = pt / (NPIX * DBINS);
  float val = prob[(size_t)n * DBINS * NPIX + (size_t)d * NPIX + pix] *
              ctx[(size_t)n * CCTX * NPIX + (size_t)c * NPIX + pix];
  atomicAdd(out + (size_t)c * 128 * 128 + v, val);
}

// ---------------- launch ----------------
extern "C" void kernel_launch(void* const* d_in, const int* in_sizes, int n_in,
                              void* d_out, int out_size, void* d_ws, size_t ws_size,
                              hipStream_t stream) {
  const float* src  = (const float*)d_in[0];
  const float* w_s1 = (const float*)d_in[1];
  const float* w_s2 = (const float*)d_in[2];
  const float* w_d1 = (const float*)d_in[3];
  const float* w_d2 = (const float*)d_in[4];
  const float* b_d  = (const float*)d_in[5];
  const float* w_c1 = (const float*)d_in[6];
  const float* w_c2 = (const float*)d_in[7];
  const float* b_c  = (const float*)d_in[8];
  const float* s2e  = (const float*)d_in[9];
  const float* intr = (const float*)d_in[10];
  const float* ida  = (const float*)d_in[11];
  const float* bda  = (const float*)d_in[12];

  char* ws = (char*)d_ws;
  __bf16* XtA  = (__bf16*)(ws + OFF_XTA);
  __bf16* XtB  = (__bf16*)(ws + OFF_XTB);
  __bf16* Ws1  = (__bf16*)(ws + OFF_WS1);
  __bf16* Ws2  = (__bf16*)(ws + OFF_WS2);
  __bf16* Wd1  = (__bf16*)(ws + OFF_WD1);
  __bf16* Wc1  = (__bf16*)(ws + OFF_WC1);
  __bf16* Wd2  = (__bf16*)(ws + OFF_WD2);
  __bf16* Wc2  = (__bf16*)(ws + OFF_WC2);
  float* logits = (float*)(ws + OFF_LOG);
  float* ctx    = (float*)(ws + OFF_CTX);
  float* cam    = (float*)(ws + OFF_CAM);
  int*   vox    = (int*)(ws + OFF_VOX);
  float* out    = (float*)d_out;

  // zero padded activation buffers (borders must be 0) and the BEV output
  {
    long n4 = (long)(2 * SZ_XT) / 16;
    k_zero_u4<<<(int)((n4 + 255) / 256), 256, 0, stream>>>((uint4*)(ws + OFF_XTA), n4);
    long no = (long)out_size / 4;  // f32 elems -> uint4
    k_zero_u4<<<(int)((no + 255) / 256), 256, 0, stream>>>((uint4*)out, no);
  }

  // weight repack / conversion
  {
    int n = KD * KD * 9, g = (n + 255) / 256;
    k_repack_w3<<<g, 256, 0, stream>>>(w_s1, Ws1);
    k_repack_w3<<<g, 256, 0, stream>>>(w_s2, Ws2);
    k_repack_w3<<<g, 256, 0, stream>>>(w_d1, Wd1);
    k_repack_w3<<<g, 256, 0, stream>>>(w_c1, Wc1);
    k_cvt_bf<<<(DBINS * KD + 255) / 256, 256, 0, stream>>>(w_d2, Wd2, DBINS * KD);
    k_cvt_bf<<<(CCTX * KD + 255) / 256, 256, 0, stream>>>(w_c2, Wc2, CCTX * KD);
  }
  k_cvt_input<<<(NCAM * CIN * NPIX + 255) / 256, 256, 0, stream>>>(src, XtA);
  k_cam_setup<<<1, 32, 0, stream>>>(s2e, intr, ida, bda, cam);

  // depthnet: four 3x3 WMMA convs + two 1x1 WMMA heads
  dim3 g3(NPIX / 32, KD / 64, NCAM);  // (22, 8, 6)
  k_conv3x3_wmma<<<g3, 32, 0, stream>>>(XtA, Ws1, XtB);   // s1
  k_conv3x3_wmma<<<g3, 32, 0, stream>>>(XtB, Ws2, XtA);   // s2
  k_conv3x3_wmma<<<g3, 32, 0, stream>>>(XtA, Wd1, XtB);   // depth branch conv
  k_conv1x1_wmma<<<dim3(NPIX / 64, DBINS / 16, NCAM), 32, 0, stream>>>(XtB, Wd2, b_d, logits, DBINS);
  k_conv3x3_wmma<<<g3, 32, 0, stream>>>(XtA, Wc1, XtB);   // context branch conv
  k_conv1x1_wmma<<<dim3(NPIX / 64, CCTX / 16, NCAM), 32, 0, stream>>>(XtB, Wc2, b_c, ctx, CCTX);

  // softmax over depth, geometry, voxel-pool scatter
  k_softmax_d<<<(NCAM * NPIX + 255) / 256, 256, 0, stream>>>(logits);
  k_geom_vox<<<(NPTS + 255) / 256, 256, 0, stream>>>(cam, vox);
  k_scatter<<<(NPTS + 2) / 3, 256, 0, stream>>>(logits, ctx, vox, out);
}
